// PointMLP_89910845375357
// MI455X (gfx1250) — matrix operand order repaired and checked by reading, not compiled, not run on hardware
//
#include <hip/hip_runtime.h>

// ---------------------------------------------------------------------------
// PointMLP forward for MI455X (gfx1250, wave32).
// Heavy matmuls: v_wmma_f32_16x16x32_f16 (f16 in, f32 accum), with the weight
// panel staged into LDS by the Tensor Data Mover (tensor_load_to_lds),
// double-buffered via s_wait_tensorcnt + workgroup barriers.
// Activations stored f16; BN stats in f32 via atomics. BN biases that are
// immediately re-centered by batch-norm are mathematically dropped.
// ---------------------------------------------------------------------------

typedef _Float16 half_t;
typedef __attribute__((ext_vector_type(8)))  half_t v8h;
typedef __attribute__((ext_vector_type(16))) half_t v16h;
typedef __attribute__((ext_vector_type(8)))  float  v8f;
typedef __attribute__((ext_vector_type(4)))  unsigned int v4u;
typedef __attribute__((ext_vector_type(8)))  unsigned int v8u;

#define BN_EPS 1e-5f
#define STD_EPS 1e-5f
#define KCH 128           // K-chunk per TDM transfer (halves)
#define LDSPAD 16         // halves of pad per LDS row (8 DWORDs via TDM pad)

// ---------------------------------------------------------------------------
// Utility kernels
// ---------------------------------------------------------------------------
__global__ void zero_kernel(float* __restrict__ p, int n) {
    int e = blockIdx.x * blockDim.x + threadIdx.x;
    if (e < n) p[e] = 0.f;
}

// Pack f32 weight (co x ci, row-major) into f16 with K padded to cip (zeros).
__global__ void pack_w_kernel(const float* __restrict__ w, half_t* __restrict__ o,
                              int co, int ci, int cip) {
    long total = (long)co * cip;
    for (long e = (long)blockIdx.x * blockDim.x + threadIdx.x; e < total;
         e += (long)gridDim.x * blockDim.x) {
        int  c = (int)(e % cip);
        long r = e / cip;
        o[e] = (c < ci) ? (half_t)w[r * ci + c] : (half_t)0.f;
    }
}

// pos0[b,m,:] = all_points[b, idx0[b,m], :]
__global__ void gather_pos_kernel(const float* __restrict__ pts,
                                  const int* __restrict__ idx0,
                                  float* __restrict__ pos0) {
    int e = blockIdx.x * blockDim.x + threadIdx.x;  // B*M0*3
    if (e >= 16 * 1024 * 3) return;
    int j = e % 3, bm = e / 3, b = bm / 1024;
    pos0[e] = pts[((long)b * 2048 + idx0[bm]) * 3 + j];
}

// Embedding matmul (CI=3, too small for WMMA): Y[row,c] = pos0[row]·emb_w[c]
__global__ void embed_kernel(const float* __restrict__ pos0,
                             const float* __restrict__ w,
                             half_t* __restrict__ Y) {
    int e = blockIdx.x * blockDim.x + threadIdx.x;  // B*M0*64
    if (e >= 16 * 1024 * 64) return;
    int c = e & 63, row = e >> 6;
    const float* p = pos0 + (long)row * 3;
    Y[e] = (half_t)(p[0] * w[c * 3 + 0] + p[1] * w[c * 3 + 1] + p[2] * w[c * 3 + 2]);
}

// ---------------------------------------------------------------------------
// BatchNorm: per-channel sum / sumsq (f32 atomics), then apply (+res, +relu)
// ---------------------------------------------------------------------------
__global__ __launch_bounds__(256)
void bn_stats_kernel(const half_t* __restrict__ Y, float* __restrict__ stats,
                     int R, int CO) {
    int c = blockIdx.x * 256 + threadIdx.x;
    if (c >= CO) return;
    long r0 = (long)blockIdx.y * 4096;
    long r1 = r0 + 4096; if (r1 > R) r1 = R;
    float s = 0.f, s2 = 0.f;
    for (long r = r0; r < r1; ++r) {
        float v = (float)Y[r * CO + c];
        s += v; s2 += v * v;
    }
    atomicAdd(&stats[c], s);
    atomicAdd(&stats[CO + c], s2);
}

__global__ __launch_bounds__(256)
void bn_apply_kernel(const half_t* __restrict__ Y, const float* __restrict__ stats,
                     const float* __restrict__ g, const float* __restrict__ b,
                     const half_t* __restrict__ resid, half_t* __restrict__ out,
                     long R, int CO, int relu) {
    long total = R * (long)CO;
    float invR = 1.f / (float)R;
    for (long e = (long)blockIdx.x * blockDim.x + threadIdx.x; e < total;
         e += (long)gridDim.x * blockDim.x) {
        int   c  = (int)(e % CO);
        float m  = stats[c] * invR;
        float vr = stats[CO + c] * invR - m * m;
        float sc = rsqrtf(vr + BN_EPS) * g[c];
        float sh = b[c] - m * sc;
        float v  = (float)Y[e] * sc + sh;
        if (resid) v += (float)resid[e];
        if (relu)  v = fmaxf(v, 0.f);
        out[e] = (half_t)v;
    }
}

// ---------------------------------------------------------------------------
// Geometric-affine pre-processing for _point_conv
// ---------------------------------------------------------------------------
__global__ __launch_bounds__(256)
void geo_stats_kernel(const float* __restrict__ posPrev, const half_t* __restrict__ featPrev,
                      const int* __restrict__ idx, float* __restrict__ mean,
                      float* __restrict__ ssd, float* __restrict__ newPos,
                      int Mp, int M, int C) {
    const int D  = C + 3;
    const int bm = blockIdx.x;            // b*M + m
    const int b  = bm / M;
    __shared__ int   sidx[32];
    __shared__ float red[256];
    const int tid = threadIdx.x;
    if (tid < 32) sidx[tid] = idx[(long)bm * 32 + tid];
    __syncthreads();
    float ssdLoc = 0.f;
    for (int c = tid; c < D; c += 256) {
        float s = 0.f, s2 = 0.f;
        for (int k = 0; k < 32; ++k) {
            int   p = sidx[k];
            float v = (c < C) ? (float)featPrev[((long)b * Mp + p) * C + c]
                              : posPrev[((long)b * Mp + p) * 3 + (c - C)];
            s += v; s2 += v * v;
        }
        float mu = s * (1.f / 32.f);
        mean[(long)bm * D + c] = mu;
        ssdLoc += s2 - s * mu;   // sum (v-mu)^2 over K
    }
    red[tid] = ssdLoc; __syncthreads();
    for (int o = 128; o > 0; o >>= 1) {
        if (tid < o) red[tid] += red[tid + o];
        __syncthreads();
    }
    if (tid == 0) atomicAdd(&ssd[b], red[0]);
    if (tid < 3) newPos[(long)bm * 3 + tid] = posPrev[((long)b * Mp + sidx[0]) * 3 + tid];
}

__global__ void std_fin_kernel(const float* __restrict__ ssd, float* __restrict__ inv,
                               int M, int D) {
    int b = threadIdx.x;
    if (b < 16) {
        float s = sqrtf(ssd[b] / (float)((long)M * 32 * D - 1));  // ddof=1
        inv[b] = 1.f / (s + STD_EPS);
    }
}

__global__ __launch_bounds__(256)
void geo_build_kernel(const float* __restrict__ posPrev, const half_t* __restrict__ featPrev,
                      const int* __restrict__ idx, const float* __restrict__ mean,
                      const float* __restrict__ inv, const float* __restrict__ alpha,
                      const float* __restrict__ beta, half_t* __restrict__ xcat,
                      int Mp, int M, int C, int CIp) {
    const int D = C + 3, CI = 2 * C + 3;
    long total = (long)16 * M * 32 * CIp;
    for (long e = (long)blockIdx.x * blockDim.x + threadIdx.x; e < total;
         e += (long)gridDim.x * blockDim.x) {
        int  c   = (int)(e % CIp);
        long row = e / CIp;
        int  k   = (int)(row % 32);
        long bm  = row / 32;
        int  b   = (int)(bm / M);
        float v;
        if (c < D) {
            int   p = idx[bm * 32 + k];
            float g = (c < C) ? (float)featPrev[((long)b * Mp + p) * C + c]
                              : posPrev[((long)b * Mp + p) * 3 + (c - C)];
            v = alpha[c] * (g - mean[bm * D + c]) * inv[b] + beta[c];
        } else if (c < CI) {
            int p0 = idx[bm * 32];  // anchor = gathered feature at k=0
            v = (float)featPrev[((long)b * Mp + p0) * C + (c - D)];
        } else {
            v = 0.f;  // K padding
        }
        xcat[e] = (half_t)v;
    }
}

// ---------------------------------------------------------------------------
// WMMA GEMM:  Y[r,n] = sum_k A[r,k] * W[n,k]   (A f16 RxKK, W f16 K-padded)
//
// Block = 4 waves = 64 rows x 64 cols. The 64x128 weight panel for the
// current K-chunk is DMA'd into LDS by the Tensor Data Mover (issued by
// wave 0 only; scalar-uniform guard), double-buffered: issue chunk i+1,
// s_wait_tensorcnt 1 => chunk i complete, barrier, compute from LDS.
// TDM pad_enable inserts 8 DWORDs per 64-col row so the LDS row stride is
// (cs+16) halves: 32B-aligned v16h fragment reads, banks spread.
// A fragments are direct 16B-aligned global loads (+global_prefetch_b8).
// ---------------------------------------------------------------------------
__global__ __launch_bounds__(128)
void wmma_gemm_kernel(const half_t* __restrict__ A, const half_t* __restrict__ W,
                      half_t* __restrict__ Y, int R, int CO, int KK, int lda) {
    __shared__ half_t bstage[2][64 * (KCH + LDSPAD)];

    const int lane  = threadIdx.x & 31;
    const int wave  = __builtin_amdgcn_readfirstlane(threadIdx.x >> 5);  // scalar
    const int nRowTiles = R >> 4;
    int rT = blockIdx.y * 4 + wave;
    const bool valid = (rT < nRowTiles);
    if (!valid) rT = nRowTiles - 1;          // clamp; skip store later
    const int colBase = blockIdx.x * 64;
    const int half    = lane >> 4;
    const int mrem    = lane & 15;
    const long row    = (long)rT * 16 + mrem;
    const half_t* arow = A + row * (long)lda;

    const int nChunks = (KK + KCH - 1) / KCH;

    // Issue one TDM descriptor: 2D tile, 64 rows (weight rows = out cols),
    // cs halves each, row stride KK halves in memory, packed to LDS with an
    // 8-DWORD pad per row. D# layout per CDNA5 ISA 8.3/8.4.
    auto issueTDM = [&](int chunk) {
        int k0 = chunk * KCH;
        int cs = KK - k0; if (cs > KCH) cs = KCH;
        // pad interval: row = cs/2 DWORDs = cs/4 qwords -> code log2(cs/4)
        unsigned intcode = (cs == KCH) ? 5u : 3u;       // 64 or 16 DWORDs/row
        unsigned lds_off = (unsigned)(size_t)(&bstage[chunk & 1][0]);  // low32 = LDS byte offset
        unsigned long long ga =
            (unsigned long long)(size_t)(W + (size_t)colBase * KK + k0);
        v4u g0 = { 1u,                                   // count=1, user mode
                   lds_off,                              // lds_addr
                   (unsigned)ga,                         // global_addr[31:0]
                   (unsigned)((ga >> 32) & 0x01FFFFFFu) | 0x80000000u }; // addr[56:32] | type=2
        v8u g1 = { 0x00010000u | (1u << 20) | (intcode << 22) | (7u << 25),
                   // ^ data_size=2B, pad_enable, pad_interval, pad_amount=8 DW
                   ((unsigned)KK & 0xFFFFu) << 16,                        // tensor_dim0 lo
                   (((unsigned)KK >> 16) & 0xFFFFu) | (((unsigned)CO & 0xFFFFu) << 16),
                   (((unsigned)CO >> 16) & 0xFFFFu) | (((unsigned)cs & 0xFFFFu) << 16),
                   64u,                                                   // tile_dim1=64
                   (unsigned)KK,                                          // dim0_stride lo32
                   0u, 0u };
        v4u gz = { 0u, 0u, 0u, 0u };                     // groups 2/3: zeros (2D)
        asm volatile("tensor_load_to_lds %0, %1, %2, %3"
                     :: "s"(g0), "s"(g1), "s"(gz), "s"(gz)
                     : "memory");
    };

    v8f acc[4];
#pragma unroll
    for (int t = 0; t < 4; ++t) acc[t] = (v8f){};

    if (wave == 0) issueTDM(0);

    for (int ic = 0; ic < nChunks; ++ic) {
        if (wave == 0) {
            if (ic + 1 < nChunks) {
                issueTDM(ic + 1);
                __builtin_amdgcn_s_wait_tensorcnt(1);   // chunk ic landed
            } else {
                __builtin_amdgcn_s_wait_tensorcnt(0);
            }
        }
        __syncthreads();                                // publish LDS to all waves

        const int k0 = ic * KCH;
        int cs = KK - k0; if (cs > KCH) cs = KCH;
        const int rstride = cs + LDSPAD;                // LDS row stride (halves)
        const half_t* bbuf = &bstage[ic & 1][0];

        for (int ks = 0; ks < cs; ks += 32) {
            // A fragment (k = ks+16*half+0..15, matching B): two 16B global loads
            v8h alo = *(const v8h*)(arow + k0 + ks + 16 * half);
            v8h ahi = *(const v8h*)(arow + k0 + ks + 16 * half + 8);
            v16h a;
#pragma unroll
            for (int j = 0; j < 8; ++j) { a[j] = alo[j]; a[j + 8] = ahi[j]; }
            if (ks + 32 >= cs && k0 + cs < KK)
                __builtin_prefetch(arow + k0 + cs, 0, 1);   // global_prefetch_b8
#pragma unroll
            for (int t = 0; t < 4; ++t) {
                // B fragment from LDS: col nl=16t+mrem, k = ks+16*half+0..15
                const half_t* bp = bbuf + (16 * t + mrem) * rstride + ks + 16 * half;
                v16h bt = *(const v16h*)bp;             // -> ds_load_b128 x2
                acc[t]  = __builtin_amdgcn_wmma_f32_16x16x32_f16(
                    false, a, false, bt, (short)0, acc[t], false, false);
            }
        }
        __syncthreads();                                // release LDS buffer
    }

    if (valid) {
#pragma unroll
        for (int t = 0; t < 4; ++t) {
#pragma unroll
            for (int r = 0; r < 8; ++r) {
                long orow = (long)rT * 16 + r + 8 * half;   // C/D layout
                Y[orow * CO + colBase + t * 16 + mrem] = (half_t)acc[t][r];
            }
        }
    }
}

// ---------------------------------------------------------------------------
// Max-pool over the middle axis of a (ROWS, NKp, CO) f16 tensor.
// ---------------------------------------------------------------------------
__global__ __launch_bounds__(256)
void maxpool_kernel(const half_t* __restrict__ in, half_t* __restrict__ out,
                    int ROWS, int NKp, int CO) {
    long total = (long)ROWS * CO;
    for (long e = (long)blockIdx.x * blockDim.x + threadIdx.x; e < total;
         e += (long)gridDim.x * blockDim.x) {
        long r = e / CO;
        int  c = (int)(e % CO);
        const half_t* p = in + r * (long)NKp * CO + c;
        float mv = -3.0e38f;
        for (int k = 0; k < NKp; ++k) mv = fmaxf(mv, (float)p[(long)k * CO]);
        out[e] = (half_t)mv;
    }
}

// Final classifier layer: out = h @ w3^T + b3   (16x256 -> 16x40, f32 out)
__global__ void cls_final_kernel(const half_t* __restrict__ h, const float* __restrict__ w3,
                                 const float* __restrict__ b3, float* __restrict__ out) {
    int e = blockIdx.x * blockDim.x + threadIdx.x;  // 16*40
    if (e >= 16 * 40) return;
    int n = e % 40, b = e / 40;
    float s = b3[n];
    for (int c = 0; c < 256; ++c) s += (float)h[b * 256 + c] * w3[n * 256 + c];
    out[e] = s;
}

// ---------------------------------------------------------------------------
// Host-side helpers
// ---------------------------------------------------------------------------
static void bn_block(const half_t* Yin, half_t* out, const half_t* resid,
                     const float* g, const float* b, long R, int CO, int relu,
                     float* stats, hipStream_t st) {
    zero_kernel<<<(2 * CO + 255) / 256, 256, 0, st>>>(stats, 2 * CO);
    dim3 gs((unsigned)((CO + 255) / 256), (unsigned)((R + 4095) / 4096));
    bn_stats_kernel<<<gs, 256, 0, st>>>(Yin, stats, (int)R, CO);
    long total = R * (long)CO;
    int  blocks = (int)((total + 255) / 256);
    bn_apply_kernel<<<blocks, 256, 0, st>>>(Yin, stats, g, b, resid, out, R, CO, relu);
}

static void gemm(const half_t* A, const half_t* W, half_t* Y, long R, int CO, int KK,
                 int lda, hipStream_t st) {
    int  rowTiles = (int)(R / 16);
    dim3 grid((unsigned)(CO / 64), (unsigned)((rowTiles + 3) / 4));
    wmma_gemm_kernel<<<grid, 128, 0, st>>>(A, W, Y, (int)R, CO, KK, lda);
}

// ---------------------------------------------------------------------------
extern "C" void kernel_launch(void* const* d_in, const int* in_sizes, int n_in,
                              void* d_out, int out_size, void* d_ws, size_t ws_size,
                              hipStream_t stream) {
    (void)in_sizes; (void)n_in; (void)out_size; (void)ws_size;

    auto F = [&](int i) { return (const float*)d_in[i]; };
    auto I = [&](int i) { return (const int*)d_in[i]; };

    // --- parameters, setup_inputs() dict (insertion) order ---
    const float* all_points = F(0);
    const int*   idx0       = I(1);
    const int*   sidxs[4]   = { I(2), I(3), I(4), I(5) };
    const float* emb_w = F(6); const float* emb_g = F(7); const float* emb_b = F(8);
    struct SP {
        const float *alpha, *beta, *w0, *g0, *b0;
        const float *rw1, *rg1, *rb1, *rw2, *rg2, *rb2;
        const float *pw1, *pg1, *pb1, *pw2, *pg2, *pb2;
    } sp[4];
    int t = 9;
    for (int s = 0; s < 4; ++s) {
        sp[s].alpha = F(t++); sp[s].beta = F(t++);
        sp[s].w0 = F(t++); sp[s].g0 = F(t++); sp[s].b0 = F(t++);
        sp[s].rw1 = F(t++); sp[s].rg1 = F(t++); sp[s].rb1 = F(t++);
        sp[s].rw2 = F(t++); sp[s].rg2 = F(t++); sp[s].rb2 = F(t++);
        sp[s].pw1 = F(t++); sp[s].pg1 = F(t++); sp[s].pb1 = F(t++);
        sp[s].pw2 = F(t++); sp[s].pg2 = F(t++); sp[s].pb2 = F(t++);
    }
    const float* cw1 = F(t++); const float* cb1 = F(t++);
    const float* cg1 = F(t++); const float* cgb1 = F(t++);
    const float* cw2 = F(t++); const float* cb2 = F(t++);
    const float* cg2 = F(t++); const float* cgb2 = F(t++);
    const float* cw3 = F(t++); const float* cb3 = F(t++);
    (void)cb1; (void)cb2;  // bias before batch-norm cancels exactly

    // --- stage geometry ---
    const int Mprev[4] = { 1024, 512, 256, 128 };
    const int Ms[4]    = { 512, 256, 128, 64 };
    const int Cin[4]   = { 64, 128, 256, 512 };
    const int Co[4]    = { 128, 256, 512, 1024 };
    int CIs[4], CIp[4];
    for (int s = 0; s < 4; ++s) { CIs[s] = 2 * Cin[s] + 3; CIp[s] = ((CIs[s] + 31) / 32) * 32; }

    // --- workspace bump allocator (256B aligned) ---
    char*  base = (char*)d_ws;
    size_t off  = 0;
    auto alloc = [&](size_t bytes) -> void* {
        off = (off + 255) & ~(size_t)255;
        void* p = base + off;
        off += bytes;
        return p;
    };

    half_t *w0p[4], *rw1p[4], *rw2p[4], *pw1p[4], *pw2p[4];
    for (int s = 0; s < 4; ++s) {
        w0p[s]  = (half_t*)alloc(sizeof(half_t) * (size_t)Co[s] * CIp[s]);
        rw1p[s] = (half_t*)alloc(sizeof(half_t) * (size_t)Co[s] * Co[s]);
        rw2p[s] = (half_t*)alloc(sizeof(half_t) * (size_t)Co[s] * Co[s]);
        pw1p[s] = (half_t*)alloc(sizeof(half_t) * (size_t)Co[s] * Co[s]);
        pw2p[s] = (half_t*)alloc(sizeof(half_t) * (size_t)Co[s] * Co[s]);
    }
    half_t* cw1p = (half_t*)alloc(sizeof(half_t) * 512 * 1024);
    half_t* cw2p = (half_t*)alloc(sizeof(half_t) * 256 * 512);

    const size_t SZ_A   = (size_t)262144 * 160;  // max R*CIp (stage0)
    const size_t SZ_ACT = (size_t)262144 * 128;  // max R*Co
    half_t* bufA = (half_t*)alloc(sizeof(half_t) * SZ_A);
    half_t* bufB = (half_t*)alloc(sizeof(half_t) * SZ_ACT);
    half_t* Y0   = (half_t*)alloc(sizeof(half_t) * SZ_ACT);
    half_t* Y1   = (half_t*)alloc(sizeof(half_t) * SZ_ACT);
    half_t* fbufA = (half_t*)alloc(sizeof(half_t) * 1048576);
    half_t* fbufB = (half_t*)alloc(sizeof(half_t) * 1048576);
    half_t* fbufM = (half_t*)alloc(sizeof(half_t) * 1048576);
    float* posA    = (float*)alloc(sizeof(float) * 16 * 1024 * 3);
    float* posB    = (float*)alloc(sizeof(float) * 16 * 1024 * 3);
    float* meanBuf = (float*)alloc(sizeof(float) * 1100000);
    float* ssd     = (float*)alloc(sizeof(float) * 16);
    float* invb    = (float*)alloc(sizeof(float) * 16);
    float* stats   = (float*)alloc(sizeof(float) * 2048);
    half_t* featC = (half_t*)alloc(sizeof(half_t) * 16 * 1024);
    half_t* yc1   = (half_t*)alloc(sizeof(half_t) * 16 * 512);
    half_t* hc1   = (half_t*)alloc(sizeof(half_t) * 16 * 512);
    half_t* yc2   = (half_t*)alloc(sizeof(half_t) * 16 * 256);
    half_t* hc2   = (half_t*)alloc(sizeof(half_t) * 16 * 256);

    // --- pack all WMMA weights to padded f16 (once per launch) ---
    auto packW = [&](const float* w, half_t* o, int co, int ci, int cip) {
        long tot = (long)co * cip;
        pack_w_kernel<<<(int)((tot + 255) / 256), 256, 0, stream>>>(w, o, co, ci, cip);
    };
    for (int s = 0; s < 4; ++s) {
        packW(sp[s].w0, w0p[s], Co[s], CIs[s], CIp[s]);
        packW(sp[s].rw1, rw1p[s], Co[s], Co[s], Co[s]);
        packW(sp[s].rw2, rw2p[s], Co[s], Co[s], Co[s]);
        packW(sp[s].pw1, pw1p[s], Co[s], Co[s], Co[s]);
        packW(sp[s].pw2, pw2p[s], Co[s], Co[s], Co[s]);
    }
    packW(cw1, cw1p, 512, 1024, 1024);
    packW(cw2, cw2p, 256, 512, 512);

    // --- embed: pos gather, 3->64 matmul, BN+relu ---
    gather_pos_kernel<<<(16 * 1024 * 3 + 255) / 256, 256, 0, stream>>>(all_points, idx0, posA);
    embed_kernel<<<(16 * 1024 * 64 + 255) / 256, 256, 0, stream>>>(posA, emb_w, Y0);
    bn_block(Y0, fbufA, nullptr, emb_g, emb_b, 16384, 64, 1, stats, stream);

    // --- 4 stages ---
    float  *posCur = posA, *posNext = posB;
    half_t *fIn = fbufA, *fOut = fbufB, *fMid = fbufM;
    for (int s = 0; s < 4; ++s) {
        const int Mp = Mprev[s], M = Ms[s], C = Cin[s], CO = Co[s], D = C + 3, cip = CIp[s];
        const long R = (long)16 * M * 32, Rp = (long)16 * M;
        const int* sidx = sidxs[s];

        // geometric affine normalization + xcat build
        zero_kernel<<<1, 32, 0, stream>>>(ssd, 16);
        geo_stats_kernel<<<16 * M, 256, 0, stream>>>(posCur, fIn, sidx, meanBuf, ssd,
                                                     posNext, Mp, M, C);
        std_fin_kernel<<<1, 16, 0, stream>>>(ssd, invb, M, D);
        {
            long tot = R * cip;
            geo_build_kernel<<<(int)((tot + 255) / 256), 256, 0, stream>>>(
                posCur, fIn, sidx, meanBuf, invb, sp[s].alpha, sp[s].beta, bufA, Mp, M, C, cip);
        }

        // conv_bn_act(w0) + res block over (B*M*K) rows
        gemm(bufA, w0p[s], Y0, R, CO, cip, cip, stream);
        bn_block(Y0, bufB, nullptr, sp[s].g0, sp[s].b0, R, CO, 1, stats, stream);   // x
        gemm(bufB, rw1p[s], Y1, R, CO, CO, CO, stream);
        bn_block(Y1, bufA, nullptr, sp[s].rg1, sp[s].rb1, R, CO, 1, stats, stream); // h
        gemm(bufA, rw2p[s], Y0, R, CO, CO, CO, stream);
        bn_block(Y0, Y1, bufB, sp[s].rg2, sp[s].rb2, R, CO, 1, stats, stream);      // relu(h+x)
        maxpool_kernel<<<(int)(((long)16 * M * CO + 255) / 256), 256, 0, stream>>>(
            Y1, fMid, 16 * M, 32, CO);

        // positional res block over (B*M) rows
        gemm(fMid, pw1p[s], Y0, Rp, CO, CO, CO, stream);
        bn_block(Y0, bufB, nullptr, sp[s].pg1, sp[s].pb1, Rp, CO, 1, stats, stream);
        gemm(bufB, pw2p[s], Y1, Rp, CO, CO, CO, stream);
        bn_block(Y1, fOut, fMid, sp[s].pg2, sp[s].pb2, Rp, CO, 1, stats, stream);

        half_t* ft = fIn; fIn = fOut; fOut = ft;
        float*  pt = posCur; posCur = posNext; posNext = pt;
    }

    // --- global max pool + classifier ---
    maxpool_kernel<<<(16 * 1024 + 255) / 256, 256, 0, stream>>>(fIn, featC, 16, 64, 1024);
    gemm(featC, cw1p, yc1, 16, 512, 1024, 1024, stream);
    bn_block(yc1, hc1, nullptr, cg1, cgb1, 16, 512, 1, stats, stream);
    gemm(hc1, cw2p, yc2, 16, 256, 512, 512, stream);
    bn_block(yc2, hc2, nullptr, cg2, cgb2, 16, 256, 1, stats, stream);
    cls_final_kernel<<<(16 * 40 + 255) / 256, 256, 0, stream>>>(hc2, cw3, cb3, (float*)d_out);
}
